// Pndb_66812511256606
// MI455X (gfx1250) — matrix operand order repaired and checked by reading, not compiled, not run on hardware
//
#include <hip/hip_runtime.h>
#include <math.h>

// ---------------- types for CDNA5 WMMA ----------------
typedef __attribute__((ext_vector_type(16))) __bf16 v16bf;
typedef __attribute__((ext_vector_type(8)))  float  v8f;

static constexpr int Dm = 512;   // model dim
static constexpr int Sq = 1024;  // seq len
static constexpr int Bb = 16;    // read batch
static constexpr int Qn = 64;    // questions
static constexpr int ND = 8;     // docs
static constexpr int NLa = 2;    // layers

enum { GF_BTRANS = 1, GF_GELU = 2, GF_BIAS = 4 };

// ================= generic batched WMMA GEMM =================
// C[M,N] = act( scale * A[M,K] @ B[K,N] + bias[N] )
// A row-major (lda). B row-major KxN (ldb) or, if GF_BTRANS, NxK row-major (ldb).
// Block tile 128x64, K step 32, 256 threads = 8 waves as 4x2; wave tile 32x32
// -> 4 accumulators / 4 WMMAs per K-step, each fragment reused twice.
#define BM 128
#define BN 64
#define BK 32

template <int FLAGS>
__global__ __launch_bounds__(256) void gemm_wmma(
    const float* __restrict__ A, const float* __restrict__ B, float* __restrict__ C,
    const float* __restrict__ bias,
    int M, int N, int K, int lda, int ldb, int ldc,
    long long sA, long long sB, long long sC,
    float scale)
{
  __shared__ __bf16 lsA[BM][BK + 2];   // pad: 17-bank row stride, conflict-free frag reads
  __shared__ __bf16 lsB[BK][BN + 4];

  const int z = blockIdx.z;
  A += (long long)z * sA;  B += (long long)z * sB;  C += (long long)z * sC;

  const int n0   = blockIdx.x * BN;
  const int m0   = blockIdx.y * BM;
  const int tid  = threadIdx.x;
  const int lane = tid & 31;
  const int wave = tid >> 5;        // 0..7
  const int wm   = wave >> 1;       // 0..3 -> M subtile *32
  const int wn   = wave & 1;        // 0..1 -> N subtile *32
  const int r    = lane & 15;
  const int h    = lane >> 4;

  v8f acc[2][2];
#pragma unroll
  for (int a = 0; a < 2; ++a)
#pragma unroll
    for (int b = 0; b < 2; ++b)
#pragma unroll
      for (int i = 0; i < 8; ++i) acc[a][b][i] = 0.f;

  for (int kk = 0; kk < K; kk += BK) {
    // ---- stage A tile 128x32 f32 -> bf16 (4 x float4 per thread) ----
#pragma unroll
    for (int t = 0; t < 4; ++t) {
      int e   = tid * 16 + t * 4;
      int row = e >> 5;
      int col = e & 31;
      int rr  = m0 + row; if (rr >= M) rr = M - 1;   // clamp reads for M < BM
      const float4 v = *(const float4*)(A + (long long)rr * lda + kk + col);
      lsA[row][col + 0] = (__bf16)v.x;
      lsA[row][col + 1] = (__bf16)v.y;
      lsA[row][col + 2] = (__bf16)v.z;
      lsA[row][col + 3] = (__bf16)v.w;
    }
    // ---- stage B tile 32x64 ----
    if (FLAGS & GF_BTRANS) {
      // B is NxK row-major: K contiguous -> float4 along K, 2 loads in flight.
#pragma unroll
      for (int t = 0; t < 2; ++t) {
        int f  = tid + t * 256;          // float4 id: 8 chunks per n-row
        int n  = f >> 3;
        int kc = (f & 7) << 2;
        const float4 v = *(const float4*)(B + (long long)(n0 + n) * ldb + kk + kc);
        lsB[kc + 0][n] = (__bf16)v.x;
        lsB[kc + 1][n] = (__bf16)v.y;
        lsB[kc + 2][n] = (__bf16)v.z;
        lsB[kc + 3][n] = (__bf16)v.w;
      }
    } else {
#pragma unroll
      for (int t = 0; t < 2; ++t) {
        int e = tid * 8 + t * 4;
        int k = e >> 6;
        int n = e & 63;
        const float4 v = *(const float4*)(B + (long long)(kk + k) * ldb + n0 + n);
        lsB[k][n + 0] = (__bf16)v.x;
        lsB[k][n + 1] = (__bf16)v.y;
        lsB[k][n + 2] = (__bf16)v.z;
        lsB[k][n + 3] = (__bf16)v.w;
      }
    }
    // prefetch next A tile (global_prefetch_b8)
    if (kk + BK < K)
      __builtin_prefetch(A + (long long)(m0 + (tid & 127)) * lda + (kk + BK) + ((tid >> 7) << 4), 0, 1);
    __syncthreads();

    // ---- build fragments per ISA 7.12.2 layouts ----
    // A 16x32 bf16: lane r holds row M=r; elems 0..7 = K h*8+i, elems 8..15 = K 16+h*8+i
    v16bf af[2], bfr[2];
#pragma unroll
    for (int a = 0; a < 2; ++a) {
      const __bf16* ap = &lsA[wm * 32 + a * 16 + r][0];
#pragma unroll
      for (int i = 0; i < 8; ++i) {
        af[a][i]     = ap[h * 8 + i];
        af[a][8 + i] = ap[16 + h * 8 + i];
      }
    }
    // B 32x16 bf16: lane r holds col N=r; elems 0..7 = K h*8+i, elems 8..15 = K 16+h*8+i
#pragma unroll
    for (int b = 0; b < 2; ++b) {
      const int bc = wn * 32 + b * 16 + r;
#pragma unroll
      for (int i = 0; i < 8; ++i) {
        bfr[b][i]     = lsB[h * 8 + i][bc];
        bfr[b][8 + i] = lsB[16 + h * 8 + i][bc];
      }
    }
#pragma unroll
    for (int a = 0; a < 2; ++a)
#pragma unroll
      for (int b = 0; b < 2; ++b)
        acc[a][b] = __builtin_amdgcn_wmma_f32_16x16x32_bf16(false, af[a], false, bfr[b],
                                                            (short)0, acc[a][b], false, false);
    __syncthreads();
  }

  // ---- epilogue: D layout VGPR v -> row v + 8h, col = lane&15 ----
#pragma unroll
  for (int a = 0; a < 2; ++a) {
    const int orow = m0 + wm * 32 + a * 16 + h * 8;
#pragma unroll
    for (int b = 0; b < 2; ++b) {
      const int col = n0 + wn * 32 + b * 16 + r;
#pragma unroll
      for (int v = 0; v < 8; ++v) {
        if (orow + v < M) {
          float val = acc[a][b][v] * scale;
          if (FLAGS & GF_BIAS) val += bias[col];
          if (FLAGS & GF_GELU) {
            float x = val;
            val = 0.5f * x * (1.f + tanhf(0.7978845608028654f * (x + 0.044715f * x * x * x)));
          }
          C[(long long)(orow + v) * ldc + col] = val;
        }
      }
    }
  }
}

// ================= softmax over rows with optional per-batch additive bias =================
__global__ void softmax_k(float* __restrict__ X, const float* __restrict__ bias,
                          int width, int rows_per_batch, int bias_stride)
{
  __shared__ float red[256];
  const int row = blockIdx.x;
  float* x = X + (long long)row * width;
  const float* bp = bias ? bias + (long long)(row / rows_per_batch) * bias_stride : nullptr;
  const int tid = threadIdx.x, bd = blockDim.x;

  float mx = -3.0e38f;
  for (int j = tid; j < width; j += bd) {
    float v = x[j] + (bp ? bp[j] : 0.f);
    x[j] = v;
    mx = fmaxf(mx, v);
  }
  red[tid] = mx; __syncthreads();
  for (int s = bd >> 1; s > 0; s >>= 1) { if (tid < s) red[tid] = fmaxf(red[tid], red[tid + s]); __syncthreads(); }
  mx = red[0]; __syncthreads();

  float sum = 0.f;
  for (int j = tid; j < width; j += bd) { float e = expf(x[j] - mx); x[j] = e; sum += e; }
  red[tid] = sum; __syncthreads();
  for (int s = bd >> 1; s > 0; s >>= 1) { if (tid < s) red[tid] += red[tid + s]; __syncthreads(); }
  const float inv = 1.f / red[0];
  for (int j = tid; j < width; j += bd) x[j] *= inv;
}

// ================= RoPE in place, one token per block (256 = D/2 pairs) =================
__global__ void rope_k(float* __restrict__ X)
{
  const int token = blockIdx.x;
  const int pos   = token & (Sq - 1);
  const int i     = threadIdx.x;                       // pair index 0..255
  const float inv = expf(-logf(10000.f) * (2.f * i) / (float)Dm);
  const float ang = (float)pos * inv;
  const float sn = sinf(ang), cs = cosf(ang);
  float* p = X + (long long)token * Dm + 2 * i;
  const float x0 = p[0], x1 = p[1];
  p[0] = x0 * cs - x1 * sn;
  p[1] = x0 * sn + x1 * cs;
}

// ================= X = LayerNorm(X + O) * g + b, one token per block =================
__global__ void ln_res_k(float* __restrict__ X, const float* __restrict__ O,
                         const float* __restrict__ g, const float* __restrict__ b)
{
  __shared__ float red[256];
  const int row = blockIdx.x, tid = threadIdx.x;
  float* x = X + (long long)row * Dm;
  const float* o = O + (long long)row * Dm;
  float v0 = x[tid] + o[tid];
  float v1 = x[tid + 256] + o[tid + 256];
  red[tid] = v0 + v1; __syncthreads();
  for (int s = 128; s > 0; s >>= 1) { if (tid < s) red[tid] += red[tid + s]; __syncthreads(); }
  const float mean = red[0] * (1.f / Dm); __syncthreads();
  const float d0 = v0 - mean, d1 = v1 - mean;
  red[tid] = d0 * d0 + d1 * d1; __syncthreads();
  for (int s = 128; s > 0; s >>= 1) { if (tid < s) red[tid] += red[tid + s]; __syncthreads(); }
  const float rstd = rsqrtf(red[0] * (1.f / Dm) + 1e-5f);
  x[tid]       = d0 * rstd * g[tid]       + b[tid];
  x[tid + 256] = d1 * rstd * g[tid + 256] + b[tid + 256];
}

// ================= y[row] = dot(X[row,:512], w) =================
__global__ void gemv512_k(const float* __restrict__ X, const float* __restrict__ w,
                          float* __restrict__ y)
{
  __shared__ float red[64];
  const int row = blockIdx.x, tid = threadIdx.x;
  const float* x = X + (long long)row * Dm;
  float s = 0.f;
#pragma unroll
  for (int t = 0; t < 8; ++t) s += x[tid + t * 64] * w[tid + t * 64];
  red[tid] = s; __syncthreads();
  for (int k = 32; k > 0; k >>= 1) { if (tid < k) red[tid] += red[tid + k]; __syncthreads(); }
  if (tid == 0) y[row] = red[0];
}

// ================= small elementwise kernels =================
__global__ void log_k(const float* __restrict__ x, float* __restrict__ y, int n)
{ int i = blockIdx.x * blockDim.x + threadIdx.x; if (i < n) y[i] = logf(x[i]); }

__global__ void logit_k(const float* __restrict__ x, float* __restrict__ y, int n)
{ int i = blockIdx.x * blockDim.x + threadIdx.x; if (i < n) y[i] = logf(x[i]) - log1pf(-x[i]); }

__global__ void gate_mul_k(const float* __restrict__ raw, const float* __restrict__ ig,
                           const float* __restrict__ b_ig, float* __restrict__ out, long long n)
{
  long long i = (long long)blockIdx.x * blockDim.x + threadIdx.x;
  if (i < n) {
    long long row = i >> 9;            // / Dm
    float s = 1.f / (1.f + expf(-(ig[row] + b_ig[0])));
    out[i] = raw[i] * s;
  }
}

__global__ void gather_selA_k(const float* __restrict__ A1s, const int* __restrict__ ids,
                              float* __restrict__ selA, long long n)
{
  long long i = (long long)blockIdx.x * blockDim.x + threadIdx.x;
  if (i < n) {
    long long per = (long long)Qn * Dm;
    int b = (int)(i / per);
    selA[i] = A1s[(long long)ids[b] * per + (i % per)];
  }
}

// out = post*(1-gate) + A2*gate ; gate written separately. One token per block.
__global__ void final_k(const float* __restrict__ post, const float* __restrict__ A2,
                        const float* __restrict__ g1, const float* __restrict__ g2,
                        const float* __restrict__ b_u1, const float* __restrict__ b_u2,
                        const float* __restrict__ b1,
                        float* __restrict__ out, float* __restrict__ gate_out)
{
  const int t = blockIdx.x;
  const float gt = 1.f / (1.f + expf(-(g1[t] + b_u1[0] + g2[t] + b_u2[0] + b1[0] - 4.f)));
  if (threadIdx.x == 0) gate_out[t] = gt;
  const long long base = (long long)t * Dm;
  for (int j = threadIdx.x; j < Dm; j += 256)
    out[base + j] = post[base + j] * (1.f - gt) + A2[base + j] * gt;
}

// ================= host-side helpers =================
static inline void launch_gemm(hipStream_t st, const float* A, const float* B, float* C,
                               const float* bias, int M, int N, int K,
                               int lda, int ldb, int ldc,
                               long long sA, long long sB, long long sC,
                               int batches, float scale, int flags)
{
  dim3 g(N / BN, (M + BM - 1) / BM, batches), b(256);
  switch (flags) {
    case 0:
      gemm_wmma<0><<<g, b, 0, st>>>(A, B, C, bias, M, N, K, lda, ldb, ldc, sA, sB, sC, scale); break;
    case GF_BIAS:
      gemm_wmma<GF_BIAS><<<g, b, 0, st>>>(A, B, C, bias, M, N, K, lda, ldb, ldc, sA, sB, sC, scale); break;
    case GF_BIAS | GF_GELU:
      gemm_wmma<GF_BIAS | GF_GELU><<<g, b, 0, st>>>(A, B, C, bias, M, N, K, lda, ldb, ldc, sA, sB, sC, scale); break;
    case GF_BTRANS:
      gemm_wmma<GF_BTRANS><<<g, b, 0, st>>>(A, B, C, bias, M, N, K, lda, ldb, ldc, sA, sB, sC, scale); break;
    default: break;
  }
}

static void run_encoder(hipStream_t st, float* X, const float* biasB, int Tb,
                        const float* Wqkv, const float* bqkv,
                        const float* Wo, const float* bo,
                        const float* ln, const float* Wff, const float* bff,
                        float* bufQ, float* bufK, float* bufV, float* bufO, float* bufS)
{
  const int T = Tb * Sq;
  const float iscale = 1.0f / sqrtf((float)Dm);
  for (int l = 0; l < NLa; ++l) {
    const float* Wq = Wqkv + (size_t)(l * 3 + 0) * Dm * Dm;
    const float* Wk = Wqkv + (size_t)(l * 3 + 1) * Dm * Dm;
    const float* Wv = Wqkv + (size_t)(l * 3 + 2) * Dm * Dm;
    launch_gemm(st, X, Wq, bufQ, bqkv + (l * 3 + 0) * Dm, T, Dm, Dm, Dm, Dm, Dm, 0, 0, 0, 1, 1.f, GF_BIAS);
    launch_gemm(st, X, Wk, bufK, bqkv + (l * 3 + 1) * Dm, T, Dm, Dm, Dm, Dm, Dm, 0, 0, 0, 1, 1.f, GF_BIAS);
    launch_gemm(st, X, Wv, bufV, bqkv + (l * 3 + 2) * Dm, T, Dm, Dm, Dm, Dm, Dm, 0, 0, 0, 1, 1.f, GF_BIAS);
    rope_k<<<T, 256, 0, st>>>(bufQ);
    rope_k<<<T, 256, 0, st>>>(bufK);
    // scores = scale * q @ k^T  (batched over Tb)
    launch_gemm(st, bufQ, bufK, bufS, nullptr, Sq, Sq, Dm, Dm, Dm, Sq,
                (long long)Sq * Dm, (long long)Sq * Dm, (long long)Sq * Sq, Tb, iscale, GF_BTRANS);
    softmax_k<<<T, 256, 0, st>>>(bufS, biasB, Sq, Sq, Sq);
    // attn @ V
    launch_gemm(st, bufS, bufV, bufO, nullptr, Sq, Dm, Sq, Sq, Dm, Dm,
                (long long)Sq * Sq, (long long)Sq * Dm, (long long)Sq * Dm, Tb, 1.f, 0);
    // o = attn_out @ Wo + bo
    launch_gemm(st, bufO, Wo + (size_t)l * Dm * Dm, bufQ, bo + l * Dm,
                T, Dm, Dm, Dm, Dm, Dm, 0, 0, 0, 1, 1.f, GF_BIAS);
    ln_res_k<<<T, 256, 0, st>>>(X, bufQ, ln + ((size_t)(l * 2 + 0) * 2 + 0) * Dm,
                                         ln + ((size_t)(l * 2 + 0) * 2 + 1) * Dm);
    // FF: gelu(X @ W1 + b1) @ W2 + b2
    launch_gemm(st, X, Wff + (size_t)(l * 2 + 0) * Dm * Dm, bufO, bff + (l * 2 + 0) * Dm,
                T, Dm, Dm, Dm, Dm, Dm, 0, 0, 0, 1, 1.f, GF_BIAS | GF_GELU);
    launch_gemm(st, bufO, Wff + (size_t)(l * 2 + 1) * Dm * Dm, bufQ, bff + (l * 2 + 1) * Dm,
                T, Dm, Dm, Dm, Dm, Dm, 0, 0, 0, 1, 1.f, GF_BIAS);
    ln_res_k<<<T, 256, 0, st>>>(X, bufQ, ln + ((size_t)(l * 2 + 1) * 2 + 0) * Dm,
                                         ln + ((size_t)(l * 2 + 1) * 2 + 1) * Dm);
  }
}

// ================= entry point =================
extern "C" void kernel_launch(void* const* d_in, const int* in_sizes, int n_in,
                              void* d_out, int out_size, void* d_ws, size_t ws_size,
                              hipStream_t stream)
{
  (void)in_sizes; (void)n_in; (void)out_size;
  const float* raw_emb   = (const float*)d_in[0];
  const float* rpw       = (const float*)d_in[1];
  const float* post_dec  = (const float*)d_in[2];
  const float* rpr       = (const float*)d_in[3];
  const float* questions = (const float*)d_in[4];
  const float* W_k       = (const float*)d_in[5];
  const float* W_ig      = (const float*)d_in[6];
  const float* b_ig      = (const float*)d_in[7];
  const float* W_u1      = (const float*)d_in[8];
  const float* b_u1      = (const float*)d_in[9];
  const float* W_u2      = (const float*)d_in[10];
  const float* b_u2      = (const float*)d_in[11];
  const float* b1s       = (const float*)d_in[12];
  const float* W_ok      = (const float*)d_in[13];
  const float* b_ok      = (const float*)d_in[14];
  const float* encw_Wqkv = (const float*)d_in[15];
  const float* encw_bqkv = (const float*)d_in[16];
  const float* encw_Wo   = (const float*)d_in[17];
  const float* encw_bo   = (const float*)d_in[18];
  const float* encw_ln   = (const float*)d_in[19];
  const float* encw_Wff  = (const float*)d_in[20];
  const float* encw_bff  = (const float*)d_in[21];
  const float* encr_Wqkv = (const float*)d_in[22];
  const float* encr_bqkv = (const float*)d_in[23];
  const float* encr_Wo   = (const float*)d_in[24];
  const float* encr_bo   = (const float*)d_in[25];
  const float* encr_ln   = (const float*)d_in[26];
  const float* encr_Wff  = (const float*)d_in[27];
  const float* encr_bff  = (const float*)d_in[28];
  const int*   lookup    = (const int*)d_in[29];

  const int Tw = ND * Sq;       // 8192 writer tokens
  const int Tr = Bb * Sq;       // 16384 reader tokens

  // ---- carve workspace (256B aligned) ----
  char* wp = (char*)d_ws;
  auto carve = [&](size_t elems) { float* p = (float*)wp; wp += (elems * 4 + 255) & ~(size_t)255; return p; };
  float* X     = carve((size_t)Tr * Dm);          // encoder x (writer then reader)
  float* bufQ  = carve((size_t)Tr * Dm);
  float* bufK  = carve((size_t)Tr * Dm);
  float* bufV  = carve((size_t)Tr * Dm);
  float* bufO  = carve((size_t)Tr * Dm);
  float* bufS  = carve((size_t)Bb * Sq * Sq);     // attention scores (max: reader)
  float* kmat  = carve((size_t)Tw * Dm);
  float* vmat  = carve((size_t)Tw * Dm);
  float* lpw   = carve((size_t)Tw);
  float* lpr   = carve((size_t)Tr);
  float* ig    = carve((size_t)Tw);
  float* S1    = carve((size_t)ND * Qn * Sq);
  float* A1    = carve((size_t)ND * Qn * Dm);
  float* selA  = carve((size_t)Bb * Qn * Dm);
  float* g1    = carve((size_t)Tr);
  float* g2    = carve((size_t)Tr);
  if ((size_t)(wp - (char*)d_ws) > ws_size) return;   // workspace too small: bail safely

  float* out_p  = (float*)d_out;
  float* gate_p = out_p + (size_t)Tr * Dm;

  const float isD = 1.0f / sqrtf((float)Dm);
  const float isQ = 1.0f / sqrtf((float)Qn);

  // ---- positional biases ----
  log_k  <<<(Tw + 255) / 256, 256, 0, stream>>>(rpw, lpw, Tw);
  logit_k<<<(Tr + 255) / 256, 256, 0, stream>>>(rpr, lpr, Tr);

  // ---- writer path ----
  launch_gemm(stream, raw_emb, W_k, kmat, nullptr, Tw, Dm, Dm, Dm, Dm, Dm, 0, 0, 0, 1, 1.f, 0);
  hipMemcpyAsync(X, raw_emb, (size_t)Tw * Dm * 4, hipMemcpyDeviceToDevice, stream);
  run_encoder(stream, X, lpw, ND, encw_Wqkv, encw_bqkv, encw_Wo, encw_bo,
              encw_ln, encw_Wff, encw_bff, bufQ, bufK, bufV, bufO, bufS);
  gemv512_k<<<Tw, 64, 0, stream>>>(X, W_ig, ig);
  gate_mul_k<<<((long long)Tw * Dm + 255) / 256, 256, 0, stream>>>(raw_emb, ig, b_ig, vmat, (long long)Tw * Dm);
  // scores1 = isD * questions @ kmat_n^T + log_pw ; softmax ; @ v
  launch_gemm(stream, questions, kmat, S1, nullptr, Qn, Sq, Dm, Dm, Dm, Sq,
              0, (long long)Sq * Dm, (long long)Qn * Sq, ND, isD, GF_BTRANS);
  softmax_k<<<ND * Qn, 256, 0, stream>>>(S1, lpw, Sq, Qn, Sq);
  launch_gemm(stream, S1, vmat, A1, nullptr, Qn, Dm, Sq, Sq, Dm, Dm,
              (long long)Qn * Sq, (long long)Sq * Dm, (long long)Qn * Dm, ND, 1.f, 0);
  gather_selA_k<<<((long long)Bb * Qn * Dm + 255) / 256, 256, 0, stream>>>(A1, lookup, selA, (long long)Bb * Qn * Dm);

  // ---- reader path ----
  hipMemcpyAsync(X, post_dec, (size_t)Tr * Dm * 4, hipMemcpyDeviceToDevice, stream);
  run_encoder(stream, X, lpr, Bb, encr_Wqkv, encr_bqkv, encr_Wo, encr_bo,
              encr_ln, encr_Wff, encr_bff, bufQ, bufK, bufV, bufO, bufS);
  // k2 = post_dec @ W_ok + b_ok (bufQ is free after encoder)
  launch_gemm(stream, post_dec, W_ok, bufQ, b_ok, Tr, Dm, Dm, Dm, Dm, Dm, 0, 0, 0, 1, 1.f, GF_BIAS);
  // a2 = softmax(isQ * k2 @ questions^T)  -> reuse bufS (Tr x Qn)
  launch_gemm(stream, bufQ, questions, bufS, nullptr, Tr, Qn, Dm, Dm, Dm, Qn,
              0, 0, 0, 1, isQ, GF_BTRANS);
  softmax_k<<<Tr, 64, 0, stream>>>(bufS, nullptr, Qn, 1, 0);
  // A2 = a2 @ selA (batched over B) -> reuse bufV
  launch_gemm(stream, bufS, selA, bufV, nullptr, Sq, Dm, Qn, Qn, Dm, Dm,
              (long long)Sq * Qn, (long long)Qn * Dm, (long long)Sq * Dm, Bb, 1.f, 0);
  // gate inputs
  gemv512_k<<<Tr, 64, 0, stream>>>(X, W_u1, g1);     // for_update @ W_u1
  gemv512_k<<<Tr, 64, 0, stream>>>(bufV, W_u2, g2);  // A2 @ W_u2
  // out + gate
  final_k<<<Tr, 256, 0, stream>>>(post_dec, bufV, g1, g2, b_u1, b_u2, b1s, out_p, gate_p);
}
